// TimePredMultiHeadAttention_39994735460879
// MI455X (gfx1250) — matrix operand order repaired and checked by reading, not compiled, not run on hardware
//
#include <hip/hip_runtime.h>

typedef _Float16 h2  __attribute__((ext_vector_type(2)));
typedef _Float16 v4h __attribute__((ext_vector_type(4)));
typedef _Float16 v8h __attribute__((ext_vector_type(8)));
typedef _Float16 v16h __attribute__((ext_vector_type(16)));
typedef float     v8f __attribute__((ext_vector_type(8)));

#define Bn   32
#define Ln   512
#define Hn   256
#define NHn  4
#define DHn  64
#define BLn  (Bn * Ln)

// Use gfx1250 async global->LDS for the attention K tiles.
#define ATTN_ASYNC_K 1

// ---------------------------------------------------------------------------
// WMMA helpers (CDNA5 v_wmma_f32_16x16x32_f16). Layouts per ISA 7.12.2:
//  A (16x32 f16): lane = half*16+lr, row=lr, pair k0 = (j>>2)*16 + half*8 + (j&3)*2
//  B (32x16 f16): col n=lr, pair k0 = (j>>2)*16 + half*8 + (j&3)*2
//  C (16x16 f32): col = lr, row = r + 8*half
// ---------------------------------------------------------------------------
__device__ __forceinline__ v8f wmma_f16(v16h a, v16h b, v8f c) {
    return __builtin_amdgcn_wmma_f32_16x16x32_f16(false, a, false, b, (short)0, c,
                                                  false, false);
}

__device__ __forceinline__ v16h frag_a_rowk(const _Float16* lane_row, int lane) {
    const int hi = lane >> 4;
    v16h a;
#pragma unroll
    for (int j = 0; j < 8; ++j) {
        const int k0 = ((j >> 2) << 4) + (hi << 3) + ((j & 3) << 1);
        h2 p = *(const h2*)(lane_row + k0);
        a[2 * j] = p.x;
        a[2 * j + 1] = p.y;
    }
    return a;
}

__device__ __forceinline__ v16h frag_b_rowk(const _Float16* lane_row, int lane) {
    const int hi = lane >> 4;
    v16h b;
#pragma unroll
    for (int j = 0; j < 8; ++j) {
        const int k0 = ((j >> 2) << 4) + (hi << 3) + ((j & 3) << 1);
        h2 p = *(const h2*)(lane_row + k0);
        b[2 * j] = p.x;
        b[2 * j + 1] = p.y;
    }
    return b;
}

__device__ __forceinline__ float redmax16(float v) {
#pragma unroll
    for (int m = 1; m <= 8; m <<= 1) v = fmaxf(v, __shfl_xor(v, m, 32));
    return v;
}
__device__ __forceinline__ float redsum16(float v) {
#pragma unroll
    for (int m = 1; m <= 8; m <<= 1) v += __shfl_xor(v, m, 32);
    return v;
}

#if ATTN_ASYNC_K
// GLOBAL_LOAD_ASYNC_TO_LDS_B128 (ISA 15.18.3 op 98): vdst = 32-bit LDS byte
// offset (low 32 bits of the generic LDS address per ISA 10.2), vaddr = 64-bit
// global address, saddr = off. Tracked by ASYNCcnt.
__device__ __forceinline__ void async_ld_lds_b128(const void* gaddr, void* lds) {
    const unsigned lo = (unsigned)(unsigned long long)lds;
    asm volatile("global_load_async_to_lds_b128 %0, %1, off" ::"v"(lo), "v"(gaddr)
                 : "memory");
}
__device__ __forceinline__ void wait_asynccnt0() {
    asm volatile("s_wait_asynccnt 0x0" ::: "memory");
}
#endif

// ---------------------------------------------------------------------------
// 0) f32 -> f16 convert (for x, used as B operand of the gate GEMM)
// ---------------------------------------------------------------------------
__global__ __launch_bounds__(256) void cvt_f16_kernel(const float* __restrict__ x,
                                                      _Float16* __restrict__ xh) {
    const int i = (blockIdx.x * 256 + threadIdx.x) * 4;
    float4 f = *(const float4*)(x + i);
    v4h o = {(_Float16)f.x, (_Float16)f.y, (_Float16)f.z, (_Float16)f.w};
    *(v4h*)(xh + i) = o;
}

// ---------------------------------------------------------------------------
// 1) Tiled WMMA GEMM:  C(M,256) = A(M,256) @ W^T + bias
//    Block tile 128x64, 8 waves (4 in M x 2 in N), wave tile 32x32 (4 WMMA).
//    Double-buffered LDS staging: one barrier per K step.
// ---------------------------------------------------------------------------
template <bool A_F16, bool OUT_F16>
__global__ __launch_bounds__(256) void gemm_xwT_bias(const void* __restrict__ A_,
                                                     const float* __restrict__ W,
                                                     const float* __restrict__ bias,
                                                     void* __restrict__ out) {
    constexpr int K = Hn;
    __shared__ _Float16 As[2][128][32];
    __shared__ _Float16 Bs[2][64][32];  // [n][k]

    const int tid = threadIdx.x;
    const int lane = tid & 31, w = tid >> 5;
    const int m0 = blockIdx.x * 128;
    const int n0 = blockIdx.y * 64;
    const int wm = w & 3, wn = w >> 2;
    const int hi = lane >> 4, lr = lane & 15;

    const int arow = tid >> 1, acs = (tid & 1) * 16;
    const int brow = tid >> 2, bks = (tid & 3) * 8;

    uint4 a16[2];
    float4 a32[4];
    float4 bf4[2];

    auto load_regs = [&](int kb) {
        if constexpr (A_F16) {
            const uint4* s4 = (const uint4*)((const _Float16*)A_ +
                                             (size_t)(m0 + arow) * K + kb + acs);
            a16[0] = s4[0];
            a16[1] = s4[1];
        } else {
            const float4* s4 =
                (const float4*)((const float*)A_ + (size_t)(m0 + arow) * K + kb + acs);
#pragma unroll
            for (int i = 0; i < 4; ++i) a32[i] = s4[i];
        }
        const float4* b4 = (const float4*)&W[(size_t)(n0 + brow) * K + kb + bks];
        bf4[0] = b4[0];
        bf4[1] = b4[1];
    };
    auto store_lds = [&](int buf) {
        if constexpr (A_F16) {
            uint4* d4 = (uint4*)&As[buf][arow][acs];
            d4[0] = a16[0];
            d4[1] = a16[1];
        } else {
#pragma unroll
            for (int i = 0; i < 4; ++i) {
                As[buf][arow][acs + 4 * i + 0] = (_Float16)a32[i].x;
                As[buf][arow][acs + 4 * i + 1] = (_Float16)a32[i].y;
                As[buf][arow][acs + 4 * i + 2] = (_Float16)a32[i].z;
                As[buf][arow][acs + 4 * i + 3] = (_Float16)a32[i].w;
            }
        }
#pragma unroll
        for (int i = 0; i < 2; ++i) {
            Bs[buf][brow][bks + 4 * i + 0] = (_Float16)bf4[i].x;
            Bs[buf][brow][bks + 4 * i + 1] = (_Float16)bf4[i].y;
            Bs[buf][brow][bks + 4 * i + 2] = (_Float16)bf4[i].z;
            Bs[buf][brow][bks + 4 * i + 3] = (_Float16)bf4[i].w;
        }
    };

    v8f c00, c01, c10, c11;
#pragma unroll
    for (int r = 0; r < 8; ++r) { c00[r] = 0.f; c01[r] = 0.f; c10[r] = 0.f; c11[r] = 0.f; }

    load_regs(0);
    store_lds(0);
    __syncthreads();

    for (int s = 0; s < K / 32; ++s) {
        const int cur = s & 1;
        if (s + 1 < K / 32) load_regs((s + 1) * 32);

        v16h a0 = frag_a_rowk(&As[cur][wm * 32 + lr][0], lane);
        v16h a1 = frag_a_rowk(&As[cur][wm * 32 + 16 + lr][0], lane);
        v16h b0 = frag_b_rowk(&Bs[cur][wn * 32 + lr][0], lane);
        v16h b1 = frag_b_rowk(&Bs[cur][wn * 32 + 16 + lr][0], lane);
        c00 = wmma_f16(a0, b0, c00);
        c01 = wmma_f16(a0, b1, c01);
        c10 = wmma_f16(a1, b0, c10);
        c11 = wmma_f16(a1, b1, c11);

        if (s + 1 < K / 32) store_lds(cur ^ 1);
        __syncthreads();
    }

    const int nA = n0 + wn * 32 + lr;
    const int nB = nA + 16;
    const float bvA = bias[nA], bvB = bias[nB];
#pragma unroll
    for (int r = 0; r < 8; ++r) {
        const int row0 = m0 + wm * 32 + r + 8 * hi;
        const int row1 = row0 + 16;
        if constexpr (OUT_F16) {
            _Float16* o = (_Float16*)out;
            o[(size_t)row0 * K + nA] = (_Float16)(c00[r] + bvA);
            o[(size_t)row0 * K + nB] = (_Float16)(c01[r] + bvB);
            o[(size_t)row1 * K + nA] = (_Float16)(c10[r] + bvA);
            o[(size_t)row1 * K + nB] = (_Float16)(c11[r] + bvB);
        } else {
            float* o = (float*)out;
            o[(size_t)row0 * K + nA] = c00[r] + bvA;
            o[(size_t)row0 * K + nB] = c01[r] + bvB;
            o[(size_t)row1 * K + nA] = c10[r] + bvA;
            o[(size_t)row1 * K + nB] = c11[r] + bvB;
        }
    }
}

// ---------------------------------------------------------------------------
// 2) Time gate:  gate[b,i,j] = sigmoid(tow1*tanh(log1p|ti-tj|*tw1+tb1)
//                                      + tow2*tanh(tq.x^T) + tob) / 8
//    tq.x^T computed with WMMA. grid = (L/128 k, L/16 q, B)
// ---------------------------------------------------------------------------
__global__ __launch_bounds__(256) void gate_kernel(
    const _Float16* __restrict__ tqh, const _Float16* __restrict__ xh,
    const float* __restrict__ time_seq, const float* __restrict__ tw1,
    const float* __restrict__ tb1, const float* __restrict__ tow1,
    const float* __restrict__ tow2, const float* __restrict__ tob,
    float* __restrict__ gate) {
    __shared__ _Float16 Atq[16][Hn];

    const int tid = threadIdx.x, lane = tid & 31, w = tid >> 5;
    const int b = blockIdx.z, q0 = blockIdx.y * 16, kb = blockIdx.x * 128 + w * 16;
    const int hi = lane >> 4, lr = lane & 15;

    {  // stage the 16 tq rows shared by all waves
        const int row = tid >> 4, cs = (tid & 15) * 16;
        const uint4* s4 =
            (const uint4*)(tqh + ((size_t)b * Ln + q0 + row) * Hn + cs);
        uint4* d4 = (uint4*)&Atq[row][cs];
        d4[0] = s4[0];
        d4[1] = s4[1];
    }
    __syncthreads();

    const int n_abs = kb + lr;
    const _Float16* xrow = xh + ((size_t)b * Ln + n_abs) * Hn;

    v8f c;
#pragma unroll
    for (int r = 0; r < 8; ++r) c[r] = 0.f;
#pragma unroll
    for (int c8 = 0; c8 < 8; ++c8) {
        v16h a = frag_a_rowk(&Atq[lr][c8 * 32], lane);
        v16h bf = frag_b_rowk(xrow + c8 * 32, lane);
        c = wmma_f16(a, bf, c);
    }

    const float tj = time_seq[b * Ln + n_abs];
#pragma unroll
    for (int r = 0; r < 8; ++r) {
        const int i = q0 + r + 8 * hi;
        const float ti = time_seq[b * Ln + i];
        const size_t idx = (size_t)i * Ln + n_abs;
        const float diff = fabsf(ti - tj);
        const float decay = tanhf(log1pf(diff) * tw1[idx] + tb1[idx]);
        const float tqk = tanhf(c[r]);
        const float g = tow1[idx] * decay + tow2[idx] * tqk + tob[idx];
        gate[(size_t)b * Ln * Ln + idx] = 0.125f / (1.f + __expf(-g));
    }
}

// ---------------------------------------------------------------------------
// 3) Fused flash attention per (b, head, 16-row q tile).
//    scores = (q.k^T)*gate + mask ; online softmax ; ctx = P.v
//    K chunk staged via async global->LDS (ASYNCcnt), V transposed through
//    registers; double-buffered so there is one barrier per chunk.
// ---------------------------------------------------------------------------
__global__ __launch_bounds__(256) void attn_kernel(
    const _Float16* __restrict__ qh, const _Float16* __restrict__ kh,
    const _Float16* __restrict__ vh, const float* __restrict__ gate,
    const float* __restrict__ mask, _Float16* __restrict__ ctxh) {
    __shared__ _Float16 ktile[2][32][DHn];
    __shared__ _Float16 vT[2][DHn][34];  // padded: kills bank-aligned strides
    __shared__ _Float16 Pbuf[8][16][32];

    const int tid = threadIdx.x, lane = tid & 31, w = tid >> 5;
    const int b = blockIdx.z, h = blockIdx.y, qb = blockIdx.x * 128 + w * 16;
    const int hi = lane >> 4, lr = lane & 15;
    const size_t rowbase = (size_t)b * Ln;

    const int srow = tid >> 3, scs = (tid & 7) * 8;  // staging coords

    // q fragments (16 rows x 64 dh), resident for the whole k loop
    const _Float16* qrow = qh + (rowbase + qb + lr) * Hn + h * DHn;
    const v16h aq0 = frag_a_rowk(qrow, lane);
    const v16h aq1 = frag_a_rowk(qrow + 32, lane);

    float m[8], l[8];
    v8f cc[4];
#pragma unroll
    for (int r = 0; r < 8; ++r) { m[r] = -1e30f; l[r] = 0.f; }
#pragma unroll
    for (int t = 0; t < 4; ++t)
#pragma unroll
        for (int r = 0; r < 8; ++r) cc[t][r] = 0.f;

    const size_t stage_off = (size_t)srow * Hn + h * DHn + scs;

    // ---- prologue: stage chunk 0 into buffer 0
    {
#if ATTN_ASYNC_K
        async_ld_lds_b128(kh + rowbase * Hn + stage_off, &ktile[0][srow][scs]);
#else
        const v8h kv = *(const v8h*)(kh + rowbase * Hn + stage_off);
        *(v8h*)&ktile[0][srow][scs] = kv;
#endif
        const v8h vv = *(const v8h*)(vh + rowbase * Hn + stage_off);
#pragma unroll
        for (int i = 0; i < 8; ++i) vT[0][scs + i][srow] = vv[i];
#if ATTN_ASYNC_K
        wait_asynccnt0();
#endif
    }
    __syncthreads();

    for (int ch = 0; ch < Ln / 32; ++ch) {
        const int cur = ch & 1, nxt = cur ^ 1;
        const int kbch = ch * 32;

        v8h vnext;
#if !ATTN_ASYNC_K
        v8h knext;
#endif
        if (ch + 1 < Ln / 32) {  // issue next-chunk staging loads now
            const size_t noff = (rowbase + (size_t)(ch + 1) * 32) * Hn + stage_off;
#if ATTN_ASYNC_K
            async_ld_lds_b128(kh + noff, &ktile[nxt][srow][scs]);
#else
            knext = *(const v8h*)(kh + noff);
#endif
            vnext = *(const v8h*)(vh + noff);
        }

        // ---- scores for this chunk
        float s0[8], s1[8];
#pragma unroll
        for (int kt2 = 0; kt2 < 2; ++kt2) {
            v8f sc;
#pragma unroll
            for (int r = 0; r < 8; ++r) sc[r] = 0.f;
            const _Float16* krow = &ktile[cur][kt2 * 16 + lr][0];
            sc = wmma_f16(aq0, frag_b_rowk(krow, lane), sc);
            sc = wmma_f16(aq1, frag_b_rowk(krow + 32, lane), sc);
            const int j = kbch + kt2 * 16 + lr;
#pragma unroll
            for (int r = 0; r < 8; ++r) {
                const int i = qb + r + 8 * hi;
                const size_t idx = (rowbase + i) * Ln + j;
                const float sv = sc[r] * gate[idx] + mask[idx];
                if (ch + 1 < Ln / 32) {
                    __builtin_prefetch(&gate[idx + 32], 0, 0);
                    __builtin_prefetch(&mask[idx + 32], 0, 0);
                }
                if (kt2 == 0) s0[r] = sv; else s1[r] = sv;
            }
        }

        // ---- online softmax (row stats live in 16-lane halves == C rows)
        float scale[8];
#pragma unroll
        for (int r = 0; r < 8; ++r) {
            const float rm = redmax16(fmaxf(s0[r], s1[r]));
            const float mn = fmaxf(m[r], rm);
            const float sc_ = __expf(m[r] - mn);
            const float p0 = __expf(s0[r] - mn);
            const float p1 = __expf(s1[r] - mn);
            l[r] = l[r] * sc_ + redsum16(p0 + p1);
            m[r] = mn;
            scale[r] = sc_;
            s0[r] = p0;
            s1[r] = p1;
        }
#pragma unroll
        for (int t = 0; t < 4; ++t)
#pragma unroll
            for (int r = 0; r < 8; ++r) cc[t][r] *= scale[r];

        // ---- C-layout probs -> A-fragment via wave-private LDS transpose
#pragma unroll
        for (int r = 0; r < 8; ++r) {
            const int prow = r + 8 * hi;
            Pbuf[w][prow][lr] = (_Float16)s0[r];
            Pbuf[w][prow][16 + lr] = (_Float16)s1[r];
        }
        const v16h ap = frag_a_rowk(&Pbuf[w][lr][0], lane);
#pragma unroll
        for (int t = 0; t < 4; ++t) {
            const v16h bv = frag_b_rowk(&vT[cur][t * 16 + lr][0], lane);
            cc[t] = wmma_f16(ap, bv, cc[t]);
        }

        // ---- finish staging of next chunk, single barrier per iteration
        if (ch + 1 < Ln / 32) {
#if !ATTN_ASYNC_K
            *(v8h*)&ktile[nxt][srow][scs] = knext;
#endif
#pragma unroll
            for (int i = 0; i < 8; ++i) vT[nxt][scs + i][srow] = vnext[i];
#if ATTN_ASYNC_K
            wait_asynccnt0();
#endif
        }
        __syncthreads();
    }

#pragma unroll
    for (int r = 0; r < 8; ++r) {
        const float inv = 1.f / l[r];
        const int row = qb + r + 8 * hi;
#pragma unroll
        for (int t = 0; t < 4; ++t)
            ctxh[(rowbase + row) * Hn + h * DHn + t * 16 + lr] =
                (_Float16)(cc[t][r] * inv);
    }
}

// ---------------------------------------------------------------------------
// 4) Residual + LayerNorm:  hs = LN(Y + x) * g + b   (one block per row)
// ---------------------------------------------------------------------------
__global__ __launch_bounds__(256) void ln_kernel(const float* __restrict__ Y,
                                                 const float* __restrict__ x,
                                                 const float* __restrict__ g,
                                                 const float* __restrict__ bb,
                                                 float* __restrict__ out) {
    __shared__ float ss[8], ssq[8];
    const size_t base = (size_t)blockIdx.x * Hn;
    const int hcol = threadIdx.x;
    const float v = Y[base + hcol] + x[base + hcol];
    float s = v, sq = v * v;
#pragma unroll
    for (int mm = 1; mm <= 16; mm <<= 1) {
        s += __shfl_xor(s, mm, 32);
        sq += __shfl_xor(sq, mm, 32);
    }
    const int w = threadIdx.x >> 5, lane = threadIdx.x & 31;
    if (lane == 0) { ss[w] = s; ssq[w] = sq; }
    __syncthreads();
    if (w == 0) {
        float a = (lane < 8) ? ss[lane] : 0.f;
        float b2 = (lane < 8) ? ssq[lane] : 0.f;
#pragma unroll
        for (int mm = 1; mm <= 4; mm <<= 1) {
            a += __shfl_xor(a, mm, 32);
            b2 += __shfl_xor(b2, mm, 32);
        }
        if (lane == 0) { ss[0] = a; ssq[0] = b2; }
    }
    __syncthreads();
    const float mu = ss[0] * (1.f / Hn);
    const float var = ssq[0] * (1.f / Hn) - mu * mu;
    const float rstd = rsqrtf(var + 1e-12f);
    out[base + hcol] = (v - mu) * rstd * g[hcol] + bb[hcol];
}

// ---------------------------------------------------------------------------
// 5) Prediction head: concat(hs[b,len-1], hs[b,len-2]) . Wt + bt
// ---------------------------------------------------------------------------
__global__ __launch_bounds__(256) void pred_kernel(const float* __restrict__ hs,
                                                   const int* __restrict__ slen,
                                                   const float* __restrict__ Wt,
                                                   const float* __restrict__ bt,
                                                   float* __restrict__ outp) {
    __shared__ float sred[8];
    const int b = blockIdx.x, tid = threadIdx.x;
    const int len = slen[b];
    float acc = 0.f;
#pragma unroll
    for (int e = tid; e < 2 * Hn; e += 256) {
        const int i = (e < Hn) ? (len - 1) : (len - 2);
        acc += hs[((size_t)b * Ln + i) * Hn + (e & (Hn - 1))] * Wt[e];
    }
#pragma unroll
    for (int mm = 1; mm <= 16; mm <<= 1) acc += __shfl_xor(acc, mm, 32);
    const int w = tid >> 5, lane = tid & 31;
    if (lane == 0) sred[w] = acc;
    __syncthreads();
    if (tid == 0) {
        float t = 0.f;
#pragma unroll
        for (int i = 0; i < 8; ++i) t += sred[i];
        outp[b] = t + bt[0];
    }
}

// ---------------------------------------------------------------------------
extern "C" void kernel_launch(void* const* d_in, const int* in_sizes, int n_in,
                              void* d_out, int out_size, void* d_ws, size_t ws_size,
                              hipStream_t stream) {
    const float* x = (const float*)d_in[0];
    const float* time_seq = (const float*)d_in[1];
    const float* attn_mask = (const float*)d_in[2];
    const int* slen = (const int*)d_in[3];
    const float* Wq = (const float*)d_in[4];
    const float* bq = (const float*)d_in[5];
    const float* Wk = (const float*)d_in[6];
    const float* bk = (const float*)d_in[7];
    const float* Wv = (const float*)d_in[8];
    const float* bv = (const float*)d_in[9];
    const float* Wd = (const float*)d_in[10];
    const float* bd = (const float*)d_in[11];
    const float* ln_g = (const float*)d_in[12];
    const float* ln_b = (const float*)d_in[13];
    const float* Wtq = (const float*)d_in[14];
    const float* btq = (const float*)d_in[15];
    const float* tw1 = (const float*)d_in[16];
    const float* tb1 = (const float*)d_in[17];
    const float* tow1 = (const float*)d_in[18];
    const float* tow2 = (const float*)d_in[19];
    const float* tob = (const float*)d_in[20];
    const float* Wt = (const float*)d_in[21];
    const float* bt = (const float*)d_in[22];

    // workspace layout
    char* ws = (char*)d_ws;
    const size_t f16mat = (size_t)BLn * Hn * sizeof(_Float16);  // 8 MB
    _Float16* xh   = (_Float16*)(ws);
    _Float16* qhp  = (_Float16*)(ws + 1 * f16mat);
    _Float16* khp  = (_Float16*)(ws + 2 * f16mat);
    _Float16* vhp  = (_Float16*)(ws + 3 * f16mat);
    _Float16* tqh  = (_Float16*)(ws + 4 * f16mat);
    _Float16* ctxh = (_Float16*)(ws + 5 * f16mat);
    float* gate = (float*)(ws + 6 * f16mat);                    // 32 MB
    float* Ybuf = (float*)(ws + 6 * f16mat + (size_t)Bn * Ln * Ln * sizeof(float));

    float* hs = (float*)d_out;
    float* pred = hs + (size_t)BLn * Hn;

    // 0) x -> f16
    cvt_f16_kernel<<<(BLn * Hn) / 1024, 256, 0, stream>>>(x, xh);

    // 1) projections q/k/v/tq (f16 out)
    dim3 ggrid(BLn / 128, Hn / 64);
    gemm_xwT_bias<false, true><<<ggrid, 256, 0, stream>>>(x, Wq, bq, qhp);
    gemm_xwT_bias<false, true><<<ggrid, 256, 0, stream>>>(x, Wk, bk, khp);
    gemm_xwT_bias<false, true><<<ggrid, 256, 0, stream>>>(x, Wv, bv, vhp);
    gemm_xwT_bias<false, true><<<ggrid, 256, 0, stream>>>(x, Wtq, btq, tqh);

    // 2) time-decay gate (includes 1/sqrt(DH))
    gate_kernel<<<dim3(Ln / 128, Ln / 16, Bn), 256, 0, stream>>>(
        tqh, xh, time_seq, tw1, tb1, tow1, tow2, tob, gate);

    // 3) fused attention -> ctx (f16)
    attn_kernel<<<dim3(Ln / 128, NHn, Bn), 256, 0, stream>>>(qhp, khp, vhp, gate,
                                                             attn_mask, ctxh);

    // 4) output projection (f32 out), then residual + LayerNorm -> hs
    gemm_xwT_bias<true, false><<<ggrid, 256, 0, stream>>>(ctxh, Wd, bd, Ybuf);
    ln_kernel<<<BLn, 256, 0, stream>>>(Ybuf, x, ln_g, ln_b, hs);

    // 5) interval prediction
    pred_kernel<<<Bn, 256, 0, stream>>>(hs, slen, Wt, bt, pred);
}
